// STAR_14199161881189
// MI455X (gfx1250) — compile-verified, matrix operand order checked by reading, not run on hardware
//
#include <hip/hip_runtime.h>

// ---------------------------------------------------------------------------
// MI455X (gfx1250) fused kernel for the 9-step STAR recurrence.
//
// Roofline: ~9.7 GFLOP total; weights ~2 MB (L2-resident, broadcast to all
// 512 workgroups); output 8.8 MB -> HBM @ 23.3 TB/s ~0.5us. Latency bound,
// so the whole recurrence is ONE kernel: h/p state lives in LDS for all 9
// steps, GEMMs use v_wmma_f32_16x16x32_bf16 (fp32 accumulate), and the
// per-row einsum (not a batch GEMM) uses VALU + per-wave LDS accumulators.
//
// New this round: a pre-pack pass converts all weight matrices fp32->bf16
// into d_ws in exact per-lane WMMA B-fragment order, so the hot loop's
// B-fragment is two contiguous 32B loads with zero conversion VALU work.
// Fallback template (ws too small) converts in-kernel as before.
// Block = 16 batch rows x 8 wave32s; grid = 8192/16 = 512 workgroups.
// ---------------------------------------------------------------------------

typedef __attribute__((ext_vector_type(16))) __bf16 v16bf;
typedef __attribute__((ext_vector_type(8)))  __bf16 v8bf;
typedef __attribute__((ext_vector_type(8)))  float  v8f;
typedef unsigned short ushort_t;

constexpr int Dd   = 40;    // hidden size
constexpr int NS   = 9;     // recurrence steps (SL-1)
constexpr int SL   = 10;
constexpr int ANN  = 268;   // output classes
constexpr int MT   = 16;    // batch rows per workgroup (one WMMA M tile)
constexpr int NWAV = 8;     // wave32s per workgroup

// packed-weight layout in d_ws (units: bf16 elements; 64 per column)
constexpr int    QW_COLS   = NS * Dd;          // 360
constexpr int    MW_COLS   = NS * Dd;          // 360
constexpr int    TW_COLS   = ANN;              // 268
constexpr int    LW_COLS   = NS * Dd * Dd;     // 14400
constexpr size_t PK_QW_OFF = 0;
constexpr size_t PK_MW_OFF = PK_QW_OFF + (size_t)QW_COLS * 64;
constexpr size_t PK_TW_OFF = PK_MW_OFF + (size_t)MW_COLS * 64;
constexpr size_t PK_LW_OFF = PK_TW_OFF + (size_t)TW_COLS * 64;
constexpr size_t PK_TOTAL  = PK_LW_OFF + (size_t)LW_COLS * 64;   // bf16 elements

__device__ __forceinline__ float relu6f(float x) { return fminf(fmaxf(x, 0.0f), 6.0f); }

__device__ __forceinline__ v8f wmma_bf16(v16bf a, v16bf b, v8f c) {
  // 8-arg form: (neg_a, A, neg_b, B, c_mod, C, reuse_a, reuse_b)
  return __builtin_amdgcn_wmma_f32_16x16x32_bf16(false, a, false, b, (short)0, c,
                                                 false, false);
}

__device__ __forceinline__ v16bf pack16(float4 a, float4 b, float4 c, float4 d) {
  v16bf r;
  r[0]  = (__bf16)a.x; r[1]  = (__bf16)a.y; r[2]  = (__bf16)a.z; r[3]  = (__bf16)a.w;
  r[4]  = (__bf16)b.x; r[5]  = (__bf16)b.y; r[6]  = (__bf16)b.z; r[7]  = (__bf16)b.w;
  r[8]  = (__bf16)c.x; r[9]  = (__bf16)c.y; r[10] = (__bf16)c.z; r[11] = (__bf16)c.w;
  r[12] = (__bf16)d.x; r[13] = (__bf16)d.y; r[14] = (__bf16)d.z; r[15] = (__bf16)d.w;
  return r;
}

// Fallback B-fragment loader from fp32 weights ([n][k] row-major, K=40).
__device__ __forceinline__ void load_bfrag_f32(const float* __restrict__ Wcol, int koff,
                                               v16bf& b0, v16bf& b1) {
  float4 x0 = *(const float4*)(Wcol + koff);
  float4 x1 = *(const float4*)(Wcol + koff + 4);
  float4 x2 = *(const float4*)(Wcol + koff + 16);
  float4 x3 = *(const float4*)(Wcol + koff + 20);
  b0 = pack16(x0, x1, x2, x3);
  float4 y0 = *(const float4*)(Wcol + 32);
  float4 y1 = *(const float4*)(Wcol + 36);
  float msk = (koff == 0) ? 1.0f : 0.0f;
  float4 z0 = make_float4(y0.x * msk, y0.y * msk, y0.z * msk, y0.w * msk);
  float4 z1 = make_float4(y1.x * msk, y1.y * msk, y1.z * msk, y1.w * msk);
  float4 zz = make_float4(0.f, 0.f, 0.f, 0.f);
  b1 = pack16(z0, z1, zz, zz);
}

// Packed B-fragment loader: per-column 64 bf16 stored in per-lane fragment
// order, so each K-chunk is one contiguous 32B load per lane.
__device__ __forceinline__ void load_bfrag_pk(const __bf16* __restrict__ col64, int lh16,
                                              v16bf& b0, v16bf& b1) {
  b0 = *(const v16bf*)(col64 + lh16);        // chunk K=0..31 for this half
  b1 = *(const v16bf*)(col64 + 32 + lh16);   // chunk K=32..63 (zero padded)
}

// A-fragment from a 64-wide bf16 LDS row (zero padded K=40..63).
__device__ __forceinline__ v16bf afrag_lds(const __bf16* rowp, int koff, int kbase) {
  v8bf r0 = *(const v8bf*)(rowp + kbase + koff);        // K = kbase+koff .. +7
  v8bf r1 = *(const v8bf*)(rowp + kbase + koff + 16);   // K = kbase+koff+16 .. +23
  v16bf a;
#pragma unroll
  for (int e = 0; e < 8; ++e) { a[e] = r0[e]; a[e + 8] = r1[e]; }
  return a;
}

// ---------------------------------------------------------------------------
// Weight pre-pack: fp32 [n][k] -> bf16 fragment-ordered [n][64].
// Per column, position groups (8 elems each) hold K bases {0,16,8,24,32,48,40,56};
// lanes then read: chunk0 @ lh*16 (K {0-7,16-23} / {8-15,24-31}),
//                  chunk1 @ 32+lh*16 (K {32-39,48-55} / {40-47,56-63}).
// One thread packs one (column, group) -> 8 contiguous bf16 (16B store).
// ---------------------------------------------------------------------------
__global__ __launch_bounds__(256)
void pack_weights(const float* __restrict__ Qw, const float* __restrict__ Mw,
                  const float* __restrict__ Tw, const float* __restrict__ Lw,
                  ushort_t* __restrict__ pk_raw)
{
  __bf16* pk = (__bf16*)pk_raw;
  int idx = blockIdx.x * blockDim.x + threadIdx.x;
  int total = (QW_COLS + MW_COLS + TW_COLS + LW_COLS) * 8;
  if (idx >= total) return;
  int group = idx & 7;
  int cg    = idx >> 3;

  const float* src; int col; size_t dst;
  if (cg < QW_COLS) {
    src = Qw; col = cg; dst = PK_QW_OFF + (size_t)cg * 64;
  } else if (cg < QW_COLS + MW_COLS) {
    src = Mw; col = cg - QW_COLS; dst = PK_MW_OFF + (size_t)col * 64;
  } else if (cg < QW_COLS + MW_COLS + TW_COLS) {
    src = Tw; col = cg - QW_COLS - MW_COLS; dst = PK_TW_OFF + (size_t)col * 64;
  } else {
    src = Lw; col = cg - QW_COLS - MW_COLS - TW_COLS; dst = PK_LW_OFF + (size_t)col * 64;
  }
  // group -> K base: {0,16,8,24,32,48,40,56} without a memory table
  int kb = ((group & 1) << 4) + ((group & 2) << 2) + ((group & 4) << 3);
  float4 a = make_float4(0.f, 0.f, 0.f, 0.f), b = a;
  if (kb < Dd) {                                   // kb in {0,8,16,24,32}; kb+7 <= 39
    a = *(const float4*)(src + (size_t)col * Dd + kb);
    b = *(const float4*)(src + (size_t)col * Dd + kb + 4);
  }
  v8bf o;
  o[0] = (__bf16)a.x; o[1] = (__bf16)a.y; o[2] = (__bf16)a.z; o[3] = (__bf16)a.w;
  o[4] = (__bf16)b.x; o[5] = (__bf16)b.y; o[6] = (__bf16)b.z; o[7] = (__bf16)b.w;
  *(v8bf*)(pk + dst + group * 8) = o;
}

// ---------------------------------------------------------------------------
// Main fused kernel (templated on whether packed weights are available).
// ---------------------------------------------------------------------------
template <bool PACKED>
__global__ __launch_bounds__(256)
void star_recurrence(const int* __restrict__ X,  const int* __restrict__ T1,
                     const int* __restrict__ T2,
                     const float* __restrict__ Ew, const float* __restrict__ Eb,
                     const float* __restrict__ Qw, const float* __restrict__ Qb,
                     const float* __restrict__ Lw, const float* __restrict__ Lb,
                     const float* __restrict__ Mw, const float* __restrict__ Mb,
                     const float* __restrict__ Tw, const float* __restrict__ Tb,
                     const float* __restrict__ emb, const ushort_t* __restrict__ pk_raw,
                     float* __restrict__ out, int Bsz)
{
  const __bf16* pk = (const __bf16*)pk_raw;

  __shared__ __align__(16) __bf16 p_bf[2][MT][64];   // p state (bf16, K-padded)
  __shared__ __align__(16) float  h_f[2][MT][48];    // h state (fp32)
  __shared__ __align__(16) __bf16 ae_bf[MT][64];     // gathered emb rows, this step
  __shared__ float t_lds[MT];
  __shared__ float ttw[NWAV][MT][Dd];                // per-wave einsum accumulators
  __shared__ float ttsum[MT][Dd];

  const int tid  = threadIdx.x;
  const int lane = tid & 31;
  const int wid  = tid >> 5;
  const int lh   = lane >> 4;       // which 16-lane half
  const int koff = lh << 3;         // K offset for this half
  const int lh16 = lh << 4;         // packed-fragment byte-group offset (elems)
  const int nlan = lane & 15;       // row (A) / column (B) index
  const int b0   = blockIdx.x * MT;

  // zero initial state (p0 = h0 = 0); K pads stay zero forever.
  for (int i = tid; i < 2 * MT * 64; i += 256) ((__bf16*)p_bf)[i] = (__bf16)0.0f;
  for (int i = tid; i < 2 * MT * 48; i += 256) ((float*)h_f)[i] = 0.0f;
  __syncthreads();

  int cur = 0;
  for (int s = 0; s < NS; ++s) {
    const int nxt = cur ^ 1;

    // ---- phase A: gather ae = emb[X[:,s,3]] and t = 8*t1+t2; zero tt accums
    for (int i = tid; i < MT * 64; i += 256) {
      int m = i >> 6, k = i & 63;
      float v = 0.0f;
      if (k < Dd) {
        int row = b0 + m; if (row >= Bsz) row = Bsz - 1;
        int id  = X[(size_t)row * SL * 4 + s * 4 + 3];
        v = emb[(size_t)id * Dd + k];
      }
      ((__bf16*)ae_bf)[i] = (__bf16)v;
    }
    if (tid < MT) {
      int row = b0 + tid; if (row >= Bsz) row = Bsz - 1;
      t_lds[tid] = (float)(T1[(size_t)row * NS + s] * 8 + T2[(size_t)row * NS + s]);
    }
    {
      float* mytt = &ttw[wid][0][0];
      for (int i = lane; i < MT * Dd; i += 32) mytt[i] = 0.0f;
    }
    __syncthreads();

    // ---- phase B: p1 = relu6(t*ew + eb + p0 @ qw^T + qb)   (3 N-tiles)
    if (wid < 3) {
      v16bf a0 = afrag_lds(&p_bf[cur][nlan][0], koff, 0);
      v16bf a1 = afrag_lds(&p_bf[cur][nlan][0], koff, 32);
      int col  = wid * 16 + nlan;
      int colc = (col < Dd) ? col : (Dd - 1);
      v16bf bb0, bb1;
      if constexpr (PACKED)
        load_bfrag_pk(pk + PK_QW_OFF + ((size_t)s * Dd + colc) * 64, lh16, bb0, bb1);
      else
        load_bfrag_f32(Qw + (size_t)s * Dd * Dd + (size_t)colc * Dd, koff, bb0, bb1);
      v8f c = {0.f, 0.f, 0.f, 0.f, 0.f, 0.f, 0.f, 0.f};
      c = wmma_bf16(a0, bb0, c);
      c = wmma_bf16(a1, bb1, c);
      float ewv = Ew[(size_t)s * Dd + colc];
      float ebv = Eb[(size_t)s * Dd + colc];
      float qbv = Qb[(size_t)s * Dd + colc];
#pragma unroll
      for (int r = 0; r < 8; ++r) {
        int m = r + (lh << 3);
        float v = relu6f(c[r] + t_lds[m] * ewv + ebv + qbv);
        if (col < Dd) p_bf[nxt][m][col] = (__bf16)v;
      }
    }
    __syncthreads();

    // ---- phase C: G = relu6(p1 @ lw^T + lb); tt[m,e] += h0[m,d]*G[m,d*40+e]
    {
      v16bf a0 = afrag_lds(&p_bf[nxt][nlan][0], koff, 0);
      v16bf a1 = afrag_lds(&p_bf[nxt][nlan][0], koff, 32);
      const float*  Lws = Lw + (size_t)s * Dd * Dd * Dd;
      const float*  Lbs = Lb + (size_t)s * Dd * Dd;
      const __bf16* Lpk = pk + PK_LW_OFF + (size_t)s * Dd * Dd * 64;
      float* mytt = &ttw[wid][0][0];
      for (int tile = wid; tile < 100; tile += NWAV) {
        int col = tile * 16 + nlan;                  // 0..1599, always in range
        v16bf bb0, bb1;
        if constexpr (PACKED) {
          if (tile + NWAV < 100)                     // warm caches for next tile
            __builtin_prefetch(Lpk + (size_t)(col + NWAV * 16) * 64, 0, 1);
          load_bfrag_pk(Lpk + (size_t)col * 64, lh16, bb0, bb1);
        } else {
          if (tile + NWAV < 100)
            __builtin_prefetch(Lws + (size_t)(col + NWAV * 16) * Dd, 0, 1);
          load_bfrag_f32(Lws + (size_t)col * Dd, koff, bb0, bb1);
        }
        v8f c = {0.f, 0.f, 0.f, 0.f, 0.f, 0.f, 0.f, 0.f};
        c = wmma_bf16(a0, bb0, c);
        c = wmma_bf16(a1, bb1, c);
        float lbv = Lbs[col];
        unsigned uc = (unsigned)col;
        int dIdx = (int)(uc / 40u);
        int eIdx = (int)(uc - (unsigned)dIdx * 40u);
#pragma unroll
        for (int r = 0; r < 8; ++r) {
          int m = r + (lh << 3);
          float g = relu6f(c[r] + lbv);
          // all (m,eIdx) targets distinct within each vector op; same-wave LDS
          // ops are ordered -> plain RMW into this wave's private accumulator.
          mytt[m * Dd + eIdx] += h_f[cur][m][dIdx] * g;
        }
      }
    }
    __syncthreads();

    // ---- reduce per-wave tt accumulators
    for (int i = tid; i < MT * Dd; i += 256) {
      float sum = 0.0f;
#pragma unroll
      for (int w = 0; w < NWAV; ++w) sum += (&ttw[w][0][0])[i];
      (&ttsum[0][0])[i] = sum;
    }
    __syncthreads();

    // ---- phase D: h1 = relu6(ae @ mw^T + mb + tt)   (3 N-tiles)
    if (wid < 3) {
      v16bf a0 = afrag_lds(&ae_bf[nlan][0], koff, 0);
      v16bf a1 = afrag_lds(&ae_bf[nlan][0], koff, 32);
      int col  = wid * 16 + nlan;
      int colc = (col < Dd) ? col : (Dd - 1);
      v16bf bb0, bb1;
      if constexpr (PACKED)
        load_bfrag_pk(pk + PK_MW_OFF + ((size_t)s * Dd + colc) * 64, lh16, bb0, bb1);
      else
        load_bfrag_f32(Mw + (size_t)s * Dd * Dd + (size_t)colc * Dd, koff, bb0, bb1);
      v8f c = {0.f, 0.f, 0.f, 0.f, 0.f, 0.f, 0.f, 0.f};
      c = wmma_bf16(a0, bb0, c);
      c = wmma_bf16(a1, bb1, c);
      float mbv = Mb[(size_t)s * Dd + colc];
#pragma unroll
      for (int r = 0; r < 8; ++r) {
        int m = r + (lh << 3);
        if (col < Dd) h_f[nxt][m][col] = relu6f(c[r] + mbv + ttsum[m][col]);
      }
    }
    __syncthreads();

    cur = nxt;
  }

  // ---- final: out = h @ Tw^T + Tb   (17 N-tiles over 8 waves)
  for (int tile = wid; tile < 17; tile += NWAV) {
    int col  = tile * 16 + nlan;
    int colc = (col < ANN) ? col : (ANN - 1);
    v16bf a0, a1;
#pragma unroll
    for (int e = 0; e < 16; ++e) {
      int k  = koff + (e & 7) + ((e >> 3) << 4);    // 0..31
      a0[e]  = (__bf16)h_f[cur][nlan][k];
      int k1 = k + 32;                               // 32..63; valid < 40
      a1[e]  = (__bf16)((k1 < Dd) ? h_f[cur][nlan][k1] : 0.0f);
    }
    v16bf bb0, bb1;
    if constexpr (PACKED)
      load_bfrag_pk(pk + PK_TW_OFF + (size_t)colc * 64, lh16, bb0, bb1);
    else
      load_bfrag_f32(Tw + (size_t)colc * Dd, koff, bb0, bb1);
    v8f c = {0.f, 0.f, 0.f, 0.f, 0.f, 0.f, 0.f, 0.f};
    c = wmma_bf16(a0, bb0, c);
    c = wmma_bf16(a1, bb1, c);
    float tbv = Tb[colc];
#pragma unroll
    for (int r = 0; r < 8; ++r) {
      int m = r + (lh << 3);
      if (col < ANN && (b0 + m) < Bsz)
        out[(size_t)(b0 + m) * ANN + col] = c[r] + tbv;
    }
  }
}

extern "C" void kernel_launch(void* const* d_in, const int* in_sizes, int n_in,
                              void* d_out, int out_size, void* d_ws, size_t ws_size,
                              hipStream_t stream) {
  const int*   X   = (const int*)d_in[0];
  const int*   T1  = (const int*)d_in[1];
  const int*   T2  = (const int*)d_in[2];
  const float* Ew  = (const float*)d_in[3];
  const float* Eb  = (const float*)d_in[4];
  const float* Qw  = (const float*)d_in[5];
  const float* Qb  = (const float*)d_in[6];
  const float* Lw  = (const float*)d_in[7];
  const float* Lb  = (const float*)d_in[8];
  const float* Mw  = (const float*)d_in[9];
  const float* Mb  = (const float*)d_in[10];
  const float* Tw  = (const float*)d_in[11];
  const float* Tb  = (const float*)d_in[12];
  const float* emb = (const float*)d_in[13];

  int Bsz  = in_sizes[0] / (SL * 4);          // 8192
  int grid = (Bsz + MT - 1) / MT;             // 512 workgroups

  const size_t need = PK_TOTAL * sizeof(ushort_t);   // ~1.97 MB
  if (ws_size >= need && d_ws != nullptr) {
    int packThreads = (QW_COLS + MW_COLS + TW_COLS + LW_COLS) * 8;
    int packGrid    = (packThreads + 255) / 256;
    hipLaunchKernelGGL(pack_weights, dim3(packGrid), dim3(256), 0, stream,
                       Qw, Mw, Tw, Lw, (ushort_t*)d_ws);
    hipLaunchKernelGGL((star_recurrence<true>), dim3(grid), dim3(256), 0, stream,
                       X, T1, T2, Ew, Eb, Qw, Qb, Lw, Lb, Mw, Mb, Tw, Tb, emb,
                       (const ushort_t*)d_ws, (float*)d_out, Bsz);
  } else {
    hipLaunchKernelGGL((star_recurrence<false>), dim3(grid), dim3(256), 0, stream,
                       X, T1, T2, Ew, Eb, Qw, Qb, Lw, Lb, Mw, Mb, Tw, Tb, emb,
                       (const ushort_t*)nullptr, (float*)d_out, Bsz);
  }
}
